// DsActor_22531398434857
// MI455X (gfx1250) — compile-verified, matrix operand order checked by reading, not run on hardware
//
#include <hip/hip_runtime.h>
#include <hip/hip_bf16.h>
#include <math.h>

// ---------------- problem constants (from reference) ----------------
#define NOBJ  5
#define DB    10
#define DO    15
#define DG    35
#define DSEM  7
#define DIN   32          // DB + DO + DSEM
#define HID   256
#define DPHI  256
#define DA    4
#define TB    16          // batch elems per workgroup
#define ROWS  (TB * NOBJ) // 80 phi-rows per workgroup

typedef __attribute__((ext_vector_type(16))) _Float16 v16h;
typedef __attribute__((ext_vector_type(8)))  _Float16 v8h;
typedef __attribute__((ext_vector_type(8)))  float    v8f;

// Packed-weight tile bases (tiles of 512 halves = one 32x16 WMMA-B tile)
#define W1_TILES   16              // phi_w1: 1 ktile x 16 ntiles
#define W2_TILES   128             // phi_w2: 8 x 16
#define VW_TILES   128             // v_w:    8 x 16
#define RW_TILES   128             // rho_w1: 8 x 16
#define QK_TILES   8               // [q_w | k_w | 0...]: 8 ktiles x 1 ntile
#define HD_TILES   8               // [mean_w | lstd_w | 0...]: 8 x 1
#define W1_OFF     0
#define W2_OFF     (W1_OFF + W1_TILES * 512)
#define VW_OFF     (W2_OFF + W2_TILES * 512)
#define RW_OFF     (VW_OFF + VW_TILES * 512)
#define QK_OFF     (RW_OFF + RW_TILES * 512)
#define HD_OFF     (QK_OFF + QK_TILES * 512)
#define PACK_TOTAL_TILES (W1_TILES + W2_TILES + VW_TILES + RW_TILES + QK_TILES + HD_TILES) // 416

// Dynamic LDS layout (bytes); every buffer 64B-aligned.
#define L_XA    0                         // 80 x 32 f16      = 5120
#define L_H1    (L_XA + ROWS*DIN*2)       // 80 x 256 f16     = 40960
#define L_H2    (L_H1 + ROWS*HID*2)       // 80 x 256 f16
#define L_VB    (L_H2 + ROWS*DPHI*2)      // 80 x 256 f16
#define L_Q     (L_VB + ROWS*DPHI*2)      // 80 f32
#define L_K     (L_Q + ROWS*4)            // 80 f32
#define L_W     (L_K + ROWS*4)            // 16 x 5 f32 = 320
#define L_POOL  (L_W + TB*NOBJ*4)         // 16 x 256 f16 (base 128960, 64B aligned)
#define L_RA    (L_POOL + TB*DPHI*2)      // 16 x 256 f16 (A-layout rho output)
#define SMEM_BYTES (L_RA + TB*HID*2)      // 145344 B

// --------------------------------------------------------------------
// Pack fp32 weights into WMMA-B fragment order:
//   tile t covers K rows [kt*32, kt*32+32), N cols [nt*16, nt*16+16).
//   packed[t*512 + lane*16 + i] = W[kt*32 + kmap(i,lane)][nt*16 + (lane&15)]
//   kmap(i,l) = i + (i>=8 ? 8 : 0) + (l>=16 ? 8 : 0)
// QK tiles:  col0 = q_w, col1 = k_w, cols 2..15 = 0
// HD tiles:  cols 0..3 = mean_w, cols 4..7 = lstd_w, cols 8..15 = 0
// --------------------------------------------------------------------
__global__ __launch_bounds__(256)
void pack_weights_kernel(const float* __restrict__ w1,
                         const float* __restrict__ w2,
                         const float* __restrict__ vw,
                         const float* __restrict__ rw,
                         const float* __restrict__ qw,
                         const float* __restrict__ kw,
                         const float* __restrict__ mw,
                         const float* __restrict__ lw,
                         _Float16* __restrict__ out) {
  int t = blockIdx.x;
  int mode = 0, kt = 0, nt = 0;
  const float* W = nullptr;
  if      (t < 16)  { W = w1; kt = 0;           nt = t; }
  else if (t < 144) { W = w2; kt = (t-16)/16;   nt = (t-16)&15; }
  else if (t < 272) { W = vw; kt = (t-144)/16;  nt = (t-144)&15; }
  else if (t < 400) { W = rw; kt = (t-272)/16;  nt = (t-272)&15; }
  else if (t < 408) { mode = 1; kt = t-400; }
  else              { mode = 2; kt = t-408; }
  int base = t * 512;

  for (int e = threadIdx.x; e < 512; e += 256) {
    int l = e >> 4, i = e & 15;
    int n = l & 15;
    int k = i + ((i >= 8) ? 8 : 0) + ((l >= 16) ? 8 : 0);
    int kr = kt*32 + k;
    float v;
    if (mode == 0)      v = W[kr * 256 + nt*16 + n];
    else if (mode == 1) v = (n == 0) ? qw[kr] : (n == 1) ? kw[kr] : 0.0f;
    else                v = (n < 4) ? mw[kr*DA + n] : (n < 8) ? lw[kr*DA + (n-4)] : 0.0f;
    out[base + e] = (_Float16)v;
  }
}

// Load an A fragment (16x32 f16) from a row-major f16 LDS matrix.
// lane (0..31): M = lane&15 (+rowbase), K block = kt*32 + (lane>=16 ? 8 : 0)
__device__ inline v16h load_a_frag(const _Float16* lds, int rowbase, int strideH,
                                   int kt, int lane) {
  int m   = lane & 15;
  int kb  = kt * 32 + ((lane >> 4) & 1) * 8;
  const _Float16* p = lds + (rowbase + m) * strideH + kb;
  v8h lo = *(const v8h*)(p);
  v8h hi = *(const v8h*)(p + 16);
  return __builtin_shufflevector(lo, hi, 0,1,2,3,4,5,6,7,8,9,10,11,12,13,14,15);
}

__global__ __launch_bounds__(256)
void actor_fused_kernel(const float* __restrict__ obs,
                        const float* __restrict__ ag,
                        const float* __restrict__ g,
                        const int*   __restrict__ sem,
                        const _Float16* __restrict__ packed,
                        const float* __restrict__ pb1,
                        const float* __restrict__ pb2,
                        const float* __restrict__ qb,
                        const float* __restrict__ kb,
                        const float* __restrict__ vb,
                        const float* __restrict__ rb1,
                        const float* __restrict__ mb,
                        const float* __restrict__ lb,
                        float* __restrict__ out, int Btot) {
  extern __shared__ char smem[];
  _Float16* xA   = (_Float16*)(smem + L_XA);
  _Float16* h1   = (_Float16*)(smem + L_H1);
  _Float16* h2   = (_Float16*)(smem + L_H2);
  _Float16* vbuf = (_Float16*)(smem + L_VB);
  float*    qls  = (float*)   (smem + L_Q);
  float*    kls  = (float*)   (smem + L_K);
  float*    wls  = (float*)   (smem + L_W);
  _Float16* pool = (_Float16*)(smem + L_POOL);
  _Float16* rA   = (_Float16*)(smem + L_RA);

  const int tid  = threadIdx.x;
  const int lane = tid & 31;
  const int wave = tid >> 5;
  const int b0   = blockIdx.x * TB;

  // Warm L2 for the packed weights this block is about to stream.
  __builtin_prefetch(packed + (size_t)tid * 832, 0, 1);

  // ---- Phase A: build x = [body | obj_n | (g-ag)[sem_ids[n]]] as f16 ----
  for (int idx = tid; idx < ROWS * DIN; idx += 256) {
    int row = idx >> 5;            // lb*5 + n
    int col = idx & 31;
    int lb  = row / NOBJ, n = row % NOBJ;
    int b   = b0 + lb;
    float v;
    if (col < DB) {
      v = obs[b * (DB + NOBJ*DO) + col];
    } else if (col < DB + DO) {
      v = obs[b * (DB + NOBJ*DO) + DB + n * DO + (col - DB)];
    } else {
      int j = col - DB - DO;
      int si = sem[n * DSEM + j];
      v = g[b * DG + si] - ag[b * DG + si];
    }
    xA[row * DIN + col] = (_Float16)v;
  }
  __syncthreads();

  const int nt0 = wave * 2, nt1 = nt0 + 1;
  const int nc0 = nt0*16 + (lane & 15), nc1 = nc0 + 16;
  const int msub = ((lane >> 4) & 1) * 8;

  // ---- Phase B: h1 = relu(x @ phi_w1 + b1)  (K=32; A shared by 2 N-tiles) ----
  {
    v16h f0 = *(const v16h*)(packed + W1_OFF + nt0*512 + lane*16);
    v16h f1 = *(const v16h*)(packed + W1_OFF + nt1*512 + lane*16);
    float bias0 = pb1[nc0], bias1 = pb1[nc1];
    for (int mt = 0; mt < 5; ++mt) {
      v16h a = load_a_frag(xA, mt * 16, DIN, 0, lane);
      v8f c0 = {}, c1 = {};
      c0 = __builtin_amdgcn_wmma_f32_16x16x32_f16(false, a, false, f0, (short)0, c0, false, false);
      c1 = __builtin_amdgcn_wmma_f32_16x16x32_f16(false, a, false, f1, (short)0, c1, false, false);
      int mrow = mt*16 + msub;
      #pragma unroll
      for (int v = 0; v < 8; ++v) {
        h1[(mrow + v) * HID + nc0] = (_Float16)fmaxf(c0[v] + bias0, 0.0f);
        h1[(mrow + v) * HID + nc1] = (_Float16)fmaxf(c1[v] + bias1, 0.0f);
      }
    }
  }
  __syncthreads();

  // ---- Phase C: h2 = relu(h1 @ phi_w2 + b2)  (B reg-resident, A shared) ----
  {
    v16h f0[8], f1[8];
    #pragma unroll
    for (int kt = 0; kt < 8; ++kt) {
      f0[kt] = *(const v16h*)(packed + W2_OFF + (kt*16 + nt0)*512 + lane*16);
      f1[kt] = *(const v16h*)(packed + W2_OFF + (kt*16 + nt1)*512 + lane*16);
    }
    float bias0 = pb2[nc0], bias1 = pb2[nc1];
    for (int mt = 0; mt < 5; ++mt) {
      v8f c0 = {}, c1 = {};
      #pragma unroll
      for (int kt = 0; kt < 8; ++kt) {
        v16h a = load_a_frag(h1, mt * 16, HID, kt, lane);
        c0 = __builtin_amdgcn_wmma_f32_16x16x32_f16(false, a, false, f0[kt], (short)0, c0, false, false);
        c1 = __builtin_amdgcn_wmma_f32_16x16x32_f16(false, a, false, f1[kt], (short)0, c1, false, false);
      }
      int mrow = mt*16 + msub;
      #pragma unroll
      for (int v = 0; v < 8; ++v) {
        h2[(mrow + v) * DPHI + nc0] = (_Float16)fmaxf(c0[v] + bias0, 0.0f);
        h2[(mrow + v) * DPHI + nc1] = (_Float16)fmaxf(c1[v] + bias1, 0.0f);
      }
    }
  }
  __syncthreads();

  // ---- Phase D: v = h2 @ v_w + v_b  (B reg-resident, A shared) ----
  {
    v16h f0[8], f1[8];
    #pragma unroll
    for (int kt = 0; kt < 8; ++kt) {
      f0[kt] = *(const v16h*)(packed + VW_OFF + (kt*16 + nt0)*512 + lane*16);
      f1[kt] = *(const v16h*)(packed + VW_OFF + (kt*16 + nt1)*512 + lane*16);
    }
    float bias0 = vb[nc0], bias1 = vb[nc1];
    for (int mt = 0; mt < 5; ++mt) {
      v8f c0 = {}, c1 = {};
      #pragma unroll
      for (int kt = 0; kt < 8; ++kt) {
        v16h a = load_a_frag(h2, mt * 16, DPHI, kt, lane);
        c0 = __builtin_amdgcn_wmma_f32_16x16x32_f16(false, a, false, f0[kt], (short)0, c0, false, false);
        c1 = __builtin_amdgcn_wmma_f32_16x16x32_f16(false, a, false, f1[kt], (short)0, c1, false, false);
      }
      int mrow = mt*16 + msub;
      #pragma unroll
      for (int v = 0; v < 8; ++v) {
        vbuf[(mrow + v) * DPHI + nc0] = (_Float16)(c0[v] + bias0);
        vbuf[(mrow + v) * DPHI + nc1] = (_Float16)(c1[v] + bias1);
      }
    }
  }
  // q/k heads on the WMMA pipe: packed [q_w | k_w | 0...] tile, waves 0..4,
  // each wave handles one M-tile of h2.  (wave-uniform branch; EXEC stays full)
  if (wave < 5) {
    int mt = wave;
    v8f c = {};
    #pragma unroll
    for (int kt = 0; kt < 8; ++kt) {
      v16h a  = load_a_frag(h2, mt * 16, DPHI, kt, lane);
      v16h fq = *(const v16h*)(packed + QK_OFF + kt*512 + lane*16);
      c = __builtin_amdgcn_wmma_f32_16x16x32_f16(false, a, false, fq, (short)0, c, false, false);
    }
    int n = lane & 15;
    int mrow = mt*16 + msub;
    if (n < 2) {
      float bias = (n == 0) ? qb[0] : kb[0];
      float* dst = (n == 0) ? qls : kls;
      #pragma unroll
      for (int v = 0; v < 8; ++v) dst[mrow + v] = c[v] + bias;
    }
  }
  __syncthreads();

  // ---- Phase F: 5x5 softmax per batch elem; column sums w[m] ----
  if (tid < TB) {
    float qv[NOBJ], kv[NOBJ], wm[NOBJ];
    #pragma unroll
    for (int n = 0; n < NOBJ; ++n) { qv[n] = qls[tid*NOBJ + n]; kv[n] = kls[tid*NOBJ + n]; wm[n] = 0.0f; }
    #pragma unroll
    for (int n = 0; n < NOBJ; ++n) {
      float s[NOBJ], mx = -3.4e38f;
      #pragma unroll
      for (int m = 0; m < NOBJ; ++m) { s[m] = qv[n] * kv[m]; mx = fmaxf(mx, s[m]); }
      float sum = 0.0f;
      #pragma unroll
      for (int m = 0; m < NOBJ; ++m) { s[m] = expf(s[m] - mx); sum += s[m]; }
      float inv = 1.0f / sum;
      #pragma unroll
      for (int m = 0; m < NOBJ; ++m) wm[m] += s[m] * inv;
    }
    #pragma unroll
    for (int m = 0; m < NOBJ; ++m) wls[tid*NOBJ + m] = wm[m];
  }
  __syncthreads();

  // ---- Phase G: pooled[b][d] = sum_m w[b][m] * v[b][m][d] ----
  {
    int d = tid;   // 256 cols, one per thread
    for (int lb = 0; lb < TB; ++lb) {
      float acc = 0.0f;
      #pragma unroll
      for (int m = 0; m < NOBJ; ++m)
        acc = fmaf(wls[lb*NOBJ + m], (float)vbuf[(lb*NOBJ + m) * DPHI + d], acc);
      pool[lb * DPHI + d] = (_Float16)acc;
    }
  }
  __syncthreads();

  // ---- Phase H: r = relu(pooled @ rho_w1 + rb1), stored f16 (A layout) ----
  {
    v8f c0 = {}, c1 = {};
    #pragma unroll
    for (int kt = 0; kt < 8; ++kt) {
      v16h a  = load_a_frag(pool, 0, DPHI, kt, lane);
      v16h g0 = *(const v16h*)(packed + RW_OFF + (kt*16 + nt0)*512 + lane*16);
      v16h g1 = *(const v16h*)(packed + RW_OFF + (kt*16 + nt1)*512 + lane*16);
      c0 = __builtin_amdgcn_wmma_f32_16x16x32_f16(false, a, false, g0, (short)0, c0, false, false);
      c1 = __builtin_amdgcn_wmma_f32_16x16x32_f16(false, a, false, g1, (short)0, c1, false, false);
    }
    float bias0 = rb1[nc0], bias1 = rb1[nc1];
    #pragma unroll
    for (int v = 0; v < 8; ++v) {
      rA[(msub + v) * HID + nc0] = (_Float16)fmaxf(c0[v] + bias0, 0.0f);
      rA[(msub + v) * HID + nc1] = (_Float16)fmaxf(c1[v] + bias1, 0.0f);
    }
  }
  __syncthreads();

  // ---- Phase I: heads on the WMMA pipe: [mean_w | lstd_w | 0...] ----
  if (wave == 0) {
    v8f c = {};
    #pragma unroll
    for (int kt = 0; kt < 8; ++kt) {
      v16h a  = load_a_frag(rA, 0, HID, kt, lane);
      v16h fh = *(const v16h*)(packed + HD_OFF + kt*512 + lane*16);
      c = __builtin_amdgcn_wmma_f32_16x16x32_f16(false, a, false, fh, (short)0, c, false, false);
    }
    int n = lane & 15;
    if (n < 8) {
      int j = n & 3;
      float bias = (n < 4) ? mb[j] : lb[j];
      #pragma unroll
      for (int v = 0; v < 8; ++v) {
        int b = msub + v;
        float acc = c[v] + bias;
        if (n < 4) {
          out[(b0 + b) * DA + j] = acc;
        } else {
          acc = fminf(fmaxf(acc, -20.0f), 2.0f);
          out[(size_t)Btot * DA + (b0 + b) * DA + j] = acc;
        }
      }
    }
  }
}

// --------------------------------------------------------------------
extern "C" void kernel_launch(void* const* d_in, const int* in_sizes, int n_in,
                              void* d_out, int out_size, void* d_ws, size_t ws_size,
                              hipStream_t stream) {
  const float* obs    = (const float*)d_in[0];
  const float* ag     = (const float*)d_in[1];
  const float* g      = (const float*)d_in[2];
  const int*   sem    = (const int*)  d_in[3];
  const float* phi_w1 = (const float*)d_in[4];
  const float* phi_b1 = (const float*)d_in[5];
  const float* phi_w2 = (const float*)d_in[6];
  const float* phi_b2 = (const float*)d_in[7];
  const float* q_w    = (const float*)d_in[8];
  const float* q_b    = (const float*)d_in[9];
  const float* k_w    = (const float*)d_in[10];
  const float* k_b    = (const float*)d_in[11];
  const float* v_w    = (const float*)d_in[12];
  const float* v_b    = (const float*)d_in[13];
  const float* rho_w1 = (const float*)d_in[14];
  const float* rho_b1 = (const float*)d_in[15];
  const float* mean_w = (const float*)d_in[16];
  const float* mean_b = (const float*)d_in[17];
  const float* lstd_w = (const float*)d_in[18];
  const float* lstd_b = (const float*)d_in[19];

  int Btot = in_sizes[0] / (DB + NOBJ * DO);   // 131072

  _Float16* packed = (_Float16*)d_ws;          // 416 tiles * 1KB = 416KB

  pack_weights_kernel<<<PACK_TOTAL_TILES, 256, 0, stream>>>(
      phi_w1, phi_w2, v_w, rho_w1, q_w, k_w, mean_w, lstd_w, packed);

  actor_fused_kernel<<<Btot / TB, 256, SMEM_BYTES, stream>>>(
      obs, ag, g, sem, packed,
      phi_b1, phi_b2, q_b, k_b, v_b, rho_b1, mean_b, lstd_b,
      (float*)d_out, Btot);
}
